// LatticeBlock_58007828300076
// MI455X (gfx1250) — matrix operand order repaired and checked by reading, not compiled, not run on hardware
//
#include <hip/hip_runtime.h>
#include <hip/hip_bf16.h>
#include <math.h>

// ---------------------------------------------------------------------------
// MI455X (gfx1250) implementation of the LatticeBlock reference.
// All large GEMMs run on v_wmma_f32_16x16x32_bf16 (bf16 inputs, f32 accum).
// wave32; 4 waves / 128 threads per GEMM block; 64-row M tiles in LDS.
// Weights pre-transposed to (Nout x K) bf16 -> every WMMA fragment is two
// contiguous ds_load_b128 per lane. 4 n-tiles per iteration (4 accumulators).
// Epilogue fully specialized at compile time (FLAGS + bias/out presence).
// ---------------------------------------------------------------------------

typedef __attribute__((ext_vector_type(16))) __bf16 v16bf;
typedef __attribute__((ext_vector_type(8)))  float  v8f;
typedef __attribute__((ext_vector_type(8)))  __bf16 bf8v;

#define H_DIM 256
#define INV_SQRT_3f 0.57735026918962576f
#define INV_SQRT_Hf 0.0625f            // 1/sqrt(256)
#define SILU_SCALEf (1.0f / 0.6f)

#define ACT_SILU 1
#define RES_ADD  2
#define RES_MUL  4

__device__ __forceinline__ __bf16 f2bf(float f) {
  unsigned u = __float_as_uint(f);
  u += 0x7fffu + ((u >> 16) & 1u);          // round-to-nearest-even
  return __builtin_bit_cast(__bf16, (unsigned short)(u >> 16));
}
__device__ __forceinline__ float ssilu(float x) {
  // x * sigmoid(x) / 0.6 with v_rcp_f32 instead of an IEEE divide
  return SILU_SCALEf * x * __builtin_amdgcn_rcpf(1.0f + __expf(-x));
}

// Two contiguous 16B LDS loads -> one 16-element bf16 fragment.
__device__ __forceinline__ v16bf frag2(const __bf16* __restrict__ p0,
                                       const __bf16* __restrict__ p1) {
  bf8v lo = *reinterpret_cast<const bf8v*>(p0);
  bf8v hi = *reinterpret_cast<const bf8v*>(p1);
  return __builtin_shufflevector(lo, hi, 0, 1, 2, 3, 4, 5, 6, 7,
                                 8, 9, 10, 11, 12, 13, 14, 15);
}

// ---- Generic tiled WMMA GEMM (fully specialized epilogue) ------------------
// C = epilogue( A(MxK,bf16,lda) @ Wt(Nout x K,bf16) + bias )
// FLAGS: ACT_SILU, RES_MUL (val*=res*alpha), RES_ADD (val+=res)
// Nout must be a multiple of 64 (4 n-tiles per iteration).
template <int K, int FLAGS, bool HAS_BIAS, bool HAS_OUTF, bool HAS_OUTBF>
__global__ __launch_bounds__(128)
void k_gemm(const __bf16* __restrict__ A, int lda,
            const __bf16* __restrict__ Wt,
            const float* __restrict__ bias,
            int M, int Nout, float alpha,
            const float* __restrict__ res, int ldr,
            float* __restrict__ outf, int ldo,
            __bf16* __restrict__ outbf, int ldob) {
  __shared__ __bf16 sA[64 * K];   // row-major M-tile
  __shared__ __bf16 sB[64 * K];   // 64 transposed weight rows (n-major, K contiguous)
  const int tid  = threadIdx.x;
  const int wave = tid >> 5, lane = tid & 31;
  const int row0 = blockIdx.x * 64;
  constexpr int Kd8 = K >> 3;

  // Stage A tile (64 x K), zero-pad rows >= M, 16B vector copies.
  for (int i = tid; i < 64 * Kd8; i += 128) {
    int r = i / Kd8, c = i - r * Kd8;
    int gr = row0 + r;
    bf8v v = {};
    if (gr < M)
      v = *reinterpret_cast<const bf8v*>(A + (size_t)gr * lda + ((size_t)c << 3));
    reinterpret_cast<bf8v*>(sA)[i] = v;
  }
  __syncthreads();

  const int half = lane >> 4;
  const int r    = lane & 15;   // A row within wave tile
  const int nloc = lane & 15;   // B column within n-tile
  const __bf16* aRow = sA + ((size_t)(wave * 16 + r)) * K;
  const bool fullTile = (row0 + 64 <= M);

  for (int nt = 0; nt < (Nout >> 4); nt += 4) {
    // Stage 4 transposed B panels: rows n = nt*16 .. nt*16+63, K contiguous.
    for (int i = tid; i < 64 * Kd8; i += 128) {
      int n = i / Kd8, c = i - n * Kd8;
      reinterpret_cast<bf8v*>(sB)[i] = *reinterpret_cast<const bf8v*>(
          Wt + (size_t)((nt << 4) + n) * K + ((size_t)c << 3));
    }
    __syncthreads();

    v8f acc[4] = {{}, {}, {}, {}};
#pragma unroll
    for (int k0 = 0; k0 < K; k0 += 32) {
      // A lane chunks: [k0+half*8, +8) and [k0+16+half*8, +8)  (AMD table order)
      v16bf af = frag2(aRow + k0 + half * 8, aRow + k0 + 16 + half * 8);
#pragma unroll
      for (int t = 0; t < 4; ++t) {
        const __bf16* bRow = sB + (size_t)(t * 16 + nloc) * K + k0;
        v16bf bfg = frag2(bRow + half * 8, bRow + 16 + half * 8);
        acc[t] = __builtin_amdgcn_wmma_f32_16x16x32_bf16(
            false, af, false, bfg, (short)0, acc[t], false, false);
      }
    }

    const int rloc = wave * 16 + half * 8;   // first of 8 result rows
#pragma unroll
    for (int t = 0; t < 4; ++t) {
      const int n = ((nt + t) << 4) + nloc;
      const float bv = HAS_BIAS ? bias[n] : 0.0f;
      if (fullTile) {
#pragma unroll
        for (int j = 0; j < 8; ++j) {
          int gr = row0 + rloc + j;
          float v = acc[t][j] + bv;
          if (FLAGS & ACT_SILU) v = ssilu(v);
          if (FLAGS & RES_MUL)  v = v * res[(size_t)gr * ldr + n] * alpha;
          if (FLAGS & RES_ADD)  v += res[(size_t)gr * ldr + n];
          if (HAS_OUTF)  outf[(size_t)gr * ldo + n] = v;
          if (HAS_OUTBF) outbf[(size_t)gr * ldob + n] = f2bf(v);
        }
      } else {
#pragma unroll
        for (int j = 0; j < 8; ++j) {
          int gr = row0 + rloc + j;
          if (gr < M) {
            float v = acc[t][j] + bv;
            if (FLAGS & ACT_SILU) v = ssilu(v);
            if (FLAGS & RES_MUL)  v = v * res[(size_t)gr * ldr + n] * alpha;
            if (FLAGS & RES_ADD)  v += res[(size_t)gr * ldr + n];
            if (HAS_OUTF)  outf[(size_t)gr * ldo + n] = v;
            if (HAS_OUTBF) outbf[(size_t)gr * ldob + n] = f2bf(v);
          }
        }
      }
    }
    __syncthreads();
  }
}

// ---- Elementwise / glue kernels --------------------------------------------
__global__ __launch_bounds__(256)
void k_copy(const float* __restrict__ s, float* __restrict__ d, size_t n) {
  size_t i = (size_t)blockIdx.x * 256 + threadIdx.x;
  if (i < n) d[i] = s[i];
}
__global__ __launch_bounds__(256)
void k_f2bf_arr(const float* __restrict__ s, __bf16* __restrict__ d, size_t n) {
  size_t i = (size_t)blockIdx.x * 256 + threadIdx.x;
  if (i < n) d[i] = f2bf(s[i]);
}
// f32 (naxes,K,Nout) -> bf16 transposed (naxes,Nout,K)
__global__ __launch_bounds__(256)
void k_f2bf_tr(const float* __restrict__ src, __bf16* __restrict__ dst,
               int K, int Nout, int naxes) {
  size_t per = (size_t)K * Nout;
  size_t i = (size_t)blockIdx.x * 256 + threadIdx.x;
  if (i >= per * naxes) return;
  int a = (int)(i / per);
  size_t rem = i - (size_t)a * per;
  int k = (int)(rem / Nout), n = (int)(rem - (size_t)k * Nout);
  dst[(size_t)a * per + (size_t)n * K + k] = f2bf(src[i]);
}

// a_s += m3 ; a_v = (m1*a_v + m2*udiff)*inv_sqrt_h   (m pre-scaled by 1/sqrt3)
__global__ __launch_bounds__(256)
void k_m_update(const float* __restrict__ m, float* __restrict__ as_f,
                __bf16* __restrict__ as_b, float* __restrict__ av_f,
                const float* __restrict__ udiff, int axis, int N) {
  size_t i = (size_t)blockIdx.x * 256 + threadIdx.x;
  if (i >= (size_t)N * H_DIM) return;
  int row = (int)(i >> 8), h = (int)(i & 255);
  size_t mb = (size_t)row * 768 + h;
  float m1 = m[mb], m2 = m[mb + 256], m3 = m[mb + 512];
  float s = as_f[i] + m3;
  as_f[i] = s;
  as_b[i] = f2bf(s);
#pragma unroll
  for (int x = 0; x < 3; ++x) {
    float ud = udiff[((size_t)row * 3 + axis) * 3 + x];
    size_t vi = ((size_t)row * 3 + x) * H_DIM + h;
    av_f[vi] = (m1 * av_f[vi] + m2 * ud) * INV_SQRT_Hf;
  }
}

// concat(a_s, s_ax[batch]) -> bf16
__global__ __launch_bounds__(256)
void k_prep_xs(const float* __restrict__ as_f, const float* __restrict__ ass,
               const int* __restrict__ batch, __bf16* __restrict__ out,
               int axis, int N) {
  size_t i = (size_t)blockIdx.x * 256 + threadIdx.x;
  if (i >= (size_t)N * 512) return;
  int row = (int)(i >> 9), c = (int)(i & 511);
  float v = (c < 256)
      ? as_f[(size_t)row * 256 + c]
      : ass[((size_t)batch[row] * 3 + axis) * 256 + (c - 256)];
  out[i] = f2bf(v);
}

// bf16(a_v + v_ax[batch]) rows = (atom,x)
__global__ __launch_bounds__(256)
void k_prep_xv(const float* __restrict__ av_f, const float* __restrict__ avs,
               const int* __restrict__ batch, __bf16* __restrict__ out,
               int axis, int N) {
  size_t i = (size_t)blockIdx.x * 256 + threadIdx.x;
  if (i >= (size_t)N * 3 * 256) return;
  int r = (int)(i >> 8), h = (int)(i & 255);
  int atom = r / 3, x = r - atom * 3;
  float v = av_f[i] +
            avs[(((size_t)batch[atom] * 3 + axis) * 3 + x) * 256 + h];
  out[i] = f2bf(v);
}

// sorted-segment mean pooling, one block per group, no atomics
__global__ __launch_bounds__(256)
void k_pool(const float* __restrict__ as_f, const float* __restrict__ av_f,
            const int* __restrict__ batch, int N,
            float* __restrict__ pooled_s, float* __restrict__ pooled_v) {
  int g = blockIdx.x, h = threadIdx.x;
  int lo = 0, hi = N;
  while (lo < hi) { int m = (lo + hi) >> 1; if (batch[m] <  g) lo = m + 1; else hi = m; }
  int start = lo; hi = N;
  while (lo < hi) { int m = (lo + hi) >> 1; if (batch[m] <= g) lo = m + 1; else hi = m; }
  int end = lo;
  int cnt = end - start;
  float denom = (float)(cnt > 0 ? cnt : 1);
  float ss = 0.f, v0 = 0.f, v1 = 0.f, v2 = 0.f;
  for (int i = start; i < end; ++i) {
    ss += as_f[(size_t)i * 256 + h];
    size_t b = (size_t)i * 768 + h;
    v0 += av_f[b]; v1 += av_f[b + 256]; v2 += av_f[b + 512];
  }
  pooled_s[(size_t)g * 256 + h] = ss / denom;
  size_t pb = (size_t)g * 768 + h;
  pooled_v[pb] = v0 / denom; pooled_v[pb + 256] = v1 / denom; pooled_v[pb + 512] = v2 / denom;
}

// (s,v) = mix(axis_scalar_state, axis_vector_state)
__global__ __launch_bounds__(256)
void k_mix1(const float* __restrict__ ass, const float* __restrict__ avs,
            const float* __restrict__ wms, const float* __restrict__ bms,
            const float* __restrict__ wmv,
            float* __restrict__ s_m, float* __restrict__ v_m, int G) {
  size_t i = (size_t)blockIdx.x * 256 + threadIdx.x;
  if (i >= (size_t)G * 256) return;
  int g = (int)(i >> 8), h = (int)(i & 255);
#pragma unroll
  for (int o = 0; o < 3; ++o) {
    float s = bms[o];
#pragma unroll
    for (int a = 0; a < 3; ++a)
      s += ass[((size_t)g * 3 + a) * 256 + h] * wms[o * 3 + a];
    s_m[((size_t)g * 3 + o) * 256 + h] = s;
#pragma unroll
    for (int x = 0; x < 3; ++x) {
      float v = 0.f;
#pragma unroll
      for (int a = 0; a < 3; ++a)
        v += avs[(((size_t)g * 3 + a) * 3 + x) * 256 + h] * wmv[o * 3 + a];
      v_m[(((size_t)g * 3 + o) * 3 + x) * 256 + h] = v;
    }
  }
}

// bf16 concat of two strided f32 row sources (256+256 cols)
__global__ __launch_bounds__(256)
void k_cat2_bf(const float* __restrict__ A, int ldaa,
               const float* __restrict__ B, int ldbb,
               __bf16* __restrict__ out, int M) {
  size_t i = (size_t)blockIdx.x * 256 + threadIdx.x;
  if (i >= (size_t)M * 512) return;
  int r = (int)(i >> 9), c = (int)(i & 511);
  float v = (c < 256) ? A[(size_t)r * ldaa + c]
                      : B[(size_t)r * ldbb + (c - 256)];
  out[i] = f2bf(v);
}

__global__ __launch_bounds__(256)
void k_prep_dv(const float* __restrict__ pooled_v, const float* __restrict__ v_m,
               __bf16* __restrict__ out, float* __restrict__ vres,
               int axis, int G) {
  size_t i = (size_t)blockIdx.x * 256 + threadIdx.x;
  if (i >= (size_t)G * 3 * 256) return;
  int r = (int)(i >> 8), h = (int)(i & 255);
  int g = r / 3, x = r - g * 3;
  float rv = v_m[(((size_t)g * 3 + axis) * 3 + x) * 256 + h];
  vres[i] = rv;
  out[i] = f2bf(pooled_v[i] + rv);
}

__global__ __launch_bounds__(256)
void k_nrm(const float* __restrict__ vp, float* __restrict__ nrm, int G) {
  size_t i = (size_t)blockIdx.x * 256 + threadIdx.x;
  if (i >= (size_t)G * 256) return;
  int g = (int)(i >> 8), h = (int)(i & 255);
  float s = 1e-8f;
#pragma unroll
  for (int x = 0; x < 3; ++x) {
    float t = vp[((size_t)(g * 3 + x)) * 512 + 256 + h];
    s += t * t;
  }
  nrm[i] = sqrtf(s);
}

__global__ __launch_bounds__(256)
void k_vnew(const float* __restrict__ s1, const float* __restrict__ vp,
            const float* __restrict__ v_tmp, float* __restrict__ v_new,
            int axis, int G) {
  size_t i = (size_t)blockIdx.x * 256 + threadIdx.x;
  if (i >= (size_t)G * 3 * 256) return;
  int r = (int)(i >> 8), h = (int)(i & 255);
  int g = r / 3, x = r - g * 3;
  v_new[(((size_t)g * 3 + axis) * 3 + x) * 256 + h] =
      s1[(size_t)g * 256 + h] * vp[(size_t)r * 512 + h] + v_tmp[i];
}

// delta[g,x,o] = sum_h ( sum_a v_new[g,a,x,h]*w_mix_v[o,a] ) * w_delta[o,h]
__global__ __launch_bounds__(256)
void k_delta(const float* __restrict__ v_new, const float* __restrict__ wmv,
             const float* __restrict__ wdelta, float* __restrict__ out, int G) {
  __shared__ float red[256 * 9];
  int g = blockIdx.x, h = threadIdx.x;
  float vv[3][3];
#pragma unroll
  for (int a = 0; a < 3; ++a)
#pragma unroll
    for (int x = 0; x < 3; ++x)
      vv[a][x] = v_new[(((size_t)g * 3 + a) * 3 + x) * 256 + h];
  float acc[9];
#pragma unroll
  for (int o = 0; o < 3; ++o) {
    float wd = wdelta[o * 256 + h];
#pragma unroll
    for (int x = 0; x < 3; ++x) {
      float m = 0.f;
#pragma unroll
      for (int a = 0; a < 3; ++a) m += vv[a][x] * wmv[o * 3 + a];
      acc[x * 3 + o] = m * wd;
    }
  }
#pragma unroll
  for (int k = 0; k < 9; ++k) red[h * 9 + k] = acc[k];
  __syncthreads();
  for (int s = 128; s > 0; s >>= 1) {
    if (h < s)
#pragma unroll
      for (int k = 0; k < 9; ++k) red[h * 9 + k] += red[(h + s) * 9 + k];
    __syncthreads();
  }
  if (h < 9) out[(size_t)g * 9 + h] = red[h];
}

// ---------------------------------------------------------------------------
extern "C" void kernel_launch(void* const* d_in, const int* in_sizes, int n_in,
                              void* d_out, int out_size, void* d_ws, size_t ws_size,
                              hipStream_t stream) {
  const int H = 256, R = 64;
  const int N = in_sizes[0] / H;
  const int G = in_sizes[2] / (3 * H);

  const float* atom_scalar  = (const float*)d_in[0];
  const float* atom_vector  = (const float*)d_in[1];
  const float* ass          = (const float*)d_in[2];   // axis_scalar_state (G,3,H)
  const float* avs          = (const float*)d_in[3];   // axis_vector_state (G,3,3,H)
  const float* lattice_feat = (const float*)d_in[4];
  const float* lattice_udiff= (const float*)d_in[5];
  const float* b_as1  = (const float*)d_in[7];
  const float* b_as2  = (const float*)d_in[9];
  const float* b_xs1  = (const float*)d_in[12];
  const float* b_xs2  = (const float*)d_in[14];
  const float* b_mp1  = (const float*)d_in[17];
  const float* b_mp2  = (const float*)d_in[19];
  const float* b_rbf  = (const float*)d_in[21];
  const float* b_ss1  = (const float*)d_in[24];
  const float* b_ss2  = (const float*)d_in[26];
  const float* w_delta= (const float*)d_in[27];
  const float* w_mix_s= (const float*)d_in[28];
  const float* b_mix_s= (const float*)d_in[29];
  const float* w_mix_v= (const float*)d_in[30];
  const int*   batch  = (const int*)  d_in[31];

  size_t off = 0;
  auto alloc = [&](size_t bytes) -> void* {
    void* p = (void*)((char*)d_ws + off);
    off += (bytes + 255) & ~(size_t)255;
    return p;
  };
  auto ew = [](size_t n) { return dim3((unsigned)((n + 255) / 256)); };
  auto gg = [](int M) { return dim3((unsigned)((M + 63) / 64)); };

  // ---- scratch layout (deterministic every call) ----
  float*  as_f   = (float*) alloc((size_t)N * H * 4);
  float*  av_f   = (float*) alloc((size_t)N * 3 * H * 4);
  __bf16* as_b   = (__bf16*)alloc((size_t)N * H * 2);
  __bf16* hid_b  = (__bf16*)alloc((size_t)N * H * 2);
  __bf16* lf_b   = (__bf16*)alloc((size_t)N * 3 * R * 2);
  float*  m_f    = (float*) alloc((size_t)N * 3 * H * 4);
  __bf16* A_big  = (__bf16*)alloc((size_t)N * 3 * H * 2);  // holds (N,512) or (3N,256)

  // transposed bf16 weights: (3, Nout, K)
  auto cvt_tr = [&](int idx, int K, int Nout) -> __bf16* {
    size_t n = (size_t)in_sizes[idx];
    __bf16* dst = (__bf16*)alloc(n * 2);
    k_f2bf_tr<<<ew(n), 256, 0, stream>>>((const float*)d_in[idx], dst, K, Nout, 3);
    return dst;
  };
  __bf16* w_as1_t = cvt_tr(6,  512, 256);
  __bf16* w_as2_t = cvt_tr(8,  256, 256);
  __bf16* w_av_t  = cvt_tr(10, 256, 256);
  __bf16* w_xs1_t = cvt_tr(11, 512, 256);
  __bf16* w_xs2_t = cvt_tr(13, 256, 256);
  __bf16* w_xv_t  = cvt_tr(15, 256, 256);
  __bf16* w_mp1_t = cvt_tr(16, 256, 256);
  __bf16* w_mp2_t = cvt_tr(18, 256, 768);
  __bf16* w_rbf_t = cvt_tr(20, 64,  768);
  __bf16* w_vp_t  = cvt_tr(22, 256, 512);
  __bf16* w_ss1_t = cvt_tr(23, 512, 256);
  __bf16* w_ss2_t = cvt_tr(25, 256, 768);

  float*  pooled_s = (float*) alloc((size_t)G * H * 4);
  float*  pooled_v = (float*) alloc((size_t)G * 3 * H * 4);
  float*  s_m      = (float*) alloc((size_t)G * 3 * H * 4);
  float*  v_m      = (float*) alloc((size_t)G * 9 * H * 4);
  __bf16* Acat     = (__bf16*)alloc((size_t)G * 512 * 2);
  __bf16* gtmp     = (__bf16*)alloc((size_t)G * H * 2);
  float*  s_tmp    = (float*) alloc((size_t)G * H * 4);
  __bf16* Adv      = (__bf16*)alloc((size_t)G * 3 * H * 2);
  float*  vres     = (float*) alloc((size_t)G * 3 * H * 4);
  float*  v_tmp    = (float*) alloc((size_t)G * 3 * H * 4);
  __bf16* v_tmp_b  = (__bf16*)alloc((size_t)G * 3 * H * 2);
  float*  vp_buf   = (float*) alloc((size_t)G * 3 * 512 * 4);
  float*  nrm      = (float*) alloc((size_t)G * H * 4);
  float*  h3s1     = (float*) alloc((size_t)G * H * 4);
  float*  v_new    = (float*) alloc((size_t)G * 9 * H * 4);

  // ---- init working copies ----
  k_copy    <<<ew((size_t)N * H),     256, 0, stream>>>(atom_scalar, as_f, (size_t)N * H);
  k_f2bf_arr<<<ew((size_t)N * H),     256, 0, stream>>>(atom_scalar, as_b, (size_t)N * H);
  k_copy    <<<ew((size_t)N * 3 * H), 256, 0, stream>>>(atom_vector, av_f, (size_t)N * 3 * H);
  k_f2bf_arr<<<ew((size_t)N * 3 * R), 256, 0, stream>>>(lattice_feat, lf_b, (size_t)N * 3 * R);

  // ==== per-atom phase: 3 chained axes ====
  for (int a = 0; a < 3; ++a) {
    // hidden = ssilu(a_s @ w_mp1 + b_mp1)
    k_gemm<256, ACT_SILU, true, false, true><<<gg(N), 128, 0, stream>>>(
        as_b, H, w_mp1_t + (size_t)a * H * H, b_mp1 + a * H,
        N, H, 0.f, nullptr, 0, nullptr, 0, hid_b, H);
    // ep = lf[:,a] @ w_rbf + b_rbf
    k_gemm<64, 0, true, true, false><<<gg(N), 128, 0, stream>>>(
        lf_b + a * R, 3 * R, w_rbf_t + (size_t)a * R * 768, b_rbf + a * 768,
        N, 768, 0.f, nullptr, 0, m_f, 768, nullptr, 0);
    // m = (hidden @ w_mp2 + b_mp2) * ep * INV_SQRT_3   (in-place on m_f)
    k_gemm<256, RES_MUL, true, true, false><<<gg(N), 128, 0, stream>>>(
        hid_b, H, w_mp2_t + (size_t)a * H * 768, b_mp2 + a * 768,
        N, 768, INV_SQRT_3f, m_f, 768, m_f, 768, nullptr, 0);
    // a_s += m3 ; a_v = (m1*a_v + m2*udiff)/sqrt(H)
    k_m_update<<<ew((size_t)N * H), 256, 0, stream>>>(m_f, as_f, as_b, av_f,
                                                      lattice_udiff, a, N);
    // xs path: a_s = mlp2(cat(a_s, s_b)) + a_s
    k_prep_xs<<<ew((size_t)N * 512), 256, 0, stream>>>(as_f, ass, batch, A_big, a, N);
    k_gemm<512, ACT_SILU, true, false, true><<<gg(N), 128, 0, stream>>>(
        A_big, 512, w_xs1_t + (size_t)a * 512 * H, b_xs1 + a * H,
        N, H, 0.f, nullptr, 0, nullptr, 0, hid_b, H);
    k_gemm<256, ACT_SILU | RES_ADD, true, true, true><<<gg(N), 128, 0, stream>>>(
        hid_b, H, w_xs2_t + (size_t)a * H * H, b_xs2 + a * H,
        N, H, 0.f, as_f, H, as_f, H, as_b, H);
    // xv path: a_v = (a_v + v_b) @ w_xv + a_v
    k_prep_xv<<<ew((size_t)N * 3 * H), 256, 0, stream>>>(av_f, avs, batch, A_big, a, N);
    k_gemm<256, RES_ADD, false, true, false><<<gg(3 * N), 128, 0, stream>>>(
        A_big, H, w_xv_t + (size_t)a * H * H, nullptr,
        3 * N, H, 0.f, av_f, H, av_f, H, nullptr, 0);
  }

  // ==== pooling (sorted batch, no atomics) ====
  k_pool<<<dim3((unsigned)G), 256, 0, stream>>>(as_f, av_f, batch, N, pooled_s, pooled_v);

  // ==== group phase ====
  k_mix1<<<ew((size_t)G * H), 256, 0, stream>>>(ass, avs, w_mix_s, b_mix_s, w_mix_v,
                                                s_m, v_m, G);
  for (int a = 0; a < 3; ++a) {
    // ds = mlp2(cat(pooled_s, s_a));  s_a += ds
    k_cat2_bf<<<ew((size_t)G * 512), 256, 0, stream>>>(pooled_s, H, s_m + a * H, 3 * H,
                                                       Acat, G);
    k_gemm<512, ACT_SILU, true, false, true><<<gg(G), 128, 0, stream>>>(
        Acat, 512, w_as1_t + (size_t)a * 512 * H, b_as1 + a * H,
        G, H, 0.f, nullptr, 0, nullptr, 0, gtmp, H);
    k_gemm<256, ACT_SILU | RES_ADD, true, true, false><<<gg(G), 128, 0, stream>>>(
        gtmp, H, w_as2_t + (size_t)a * H * H, b_as2 + a * H,
        G, H, 0.f, s_m + a * H, 3 * H, s_tmp, H, nullptr, 0);
    // dv = (pooled_v + v_a) @ w_av ;  v_a += dv
    k_prep_dv<<<ew((size_t)G * 3 * H), 256, 0, stream>>>(pooled_v, v_m, Adv, vres, a, G);
    k_gemm<256, RES_ADD, false, true, true><<<gg(3 * G), 128, 0, stream>>>(
        Adv, H, w_av_t + (size_t)a * H * H, nullptr,
        3 * G, H, 0.f, vres, H, v_tmp, H, v_tmp_b, H);
    // vp = v_a @ w_vp  -> v1|v2
    k_gemm<256, 0, false, true, false><<<gg(3 * G), 128, 0, stream>>>(
        v_tmp_b, H, w_vp_t + (size_t)a * H * 512, nullptr,
        3 * G, 512, 0.f, nullptr, 0, vp_buf, 512, nullptr, 0);
    k_nrm<<<ew((size_t)G * H), 256, 0, stream>>>(vp_buf, nrm, G);
    // s1 = first third of (ssilu(cat(s_a, nrm) @ w_ss1 + b) @ w_ss2 + b)
    k_cat2_bf<<<ew((size_t)G * 512), 256, 0, stream>>>(s_tmp, H, nrm, H, Acat, G);
    k_gemm<512, ACT_SILU, true, false, true><<<gg(G), 128, 0, stream>>>(
        Acat, 512, w_ss1_t + (size_t)a * 512 * H, b_ss1 + a * H,
        G, H, 0.f, nullptr, 0, nullptr, 0, gtmp, H);
    // transposed w_ss2: rows n=0..255 give the s1 third
    k_gemm<256, 0, true, true, false><<<gg(G), 128, 0, stream>>>(
        gtmp, H, w_ss2_t + (size_t)a * H * 768, b_ss2 + a * 768,
        G, H, 0.f, nullptr, 0, h3s1, H, nullptr, 0);
    // v_new = s1*v1 + v_a
    k_vnew<<<ew((size_t)G * 3 * H), 256, 0, stream>>>(h3s1, vp_buf, v_tmp, v_new, a, G);
  }

  // ==== final mix + delta ====
  k_delta<<<dim3((unsigned)G), 256, 0, stream>>>(v_new, w_mix_v, w_delta,
                                                 (float*)d_out, G);

  (void)n_in; (void)out_size; (void)ws_size;
}
